// Attention_31112743092855
// MI455X (gfx1250) — compile-verified, matrix operand order checked by reading, not run on hardware
//
#include <hip/hip_runtime.h>

#define DIM    512
#define HEADS  8
#define DHEAD  64
#define NTOK   4096
#define BATCH  2
#define ROWS   (BATCH * NTOK)      // 8192
#define QKVN   (3 * DIM)           // 1536
#define SCALE  0.125f              // 64^-0.5
#define KSTEP  32                  // keys per flash stage

typedef __attribute__((ext_vector_type(16))) _Float16     v16h;
typedef __attribute__((ext_vector_type(8)))  _Float16     v8h;
typedef __attribute__((ext_vector_type(8)))  float        v8f;
typedef __attribute__((ext_vector_type(4)))  unsigned int u32x4;
typedef __attribute__((ext_vector_type(8)))  unsigned int u32x8;

// ---------------------------------------------------------------------------
// A/B fragment loader for V_WMMA_F32_16X16X32_F16.
// p = start of this lane's 32-element K run (row of A for M = lane%16,
// column of B for N = lane%16, contiguous along K). Per ISA layout:
// lanes 0-15 hold K=[0..7]+[16..23]; lanes 16-31 hold K=[8..15]+[24..31].
// ---------------------------------------------------------------------------
__device__ __forceinline__ v16h load_frag(const _Float16* p) {
    int lane = threadIdx.x & 31;
    int kb = (lane & 16) ? 8 : 0;
    v8h lo = *(const v8h*)(p + kb);
    v8h hi = *(const v8h*)(p + kb + 16);
    v16h r;
#pragma unroll
    for (int i = 0; i < 8; ++i) { r[i] = lo[i]; r[8 + i] = hi[i]; }
    return r;
}

__device__ __forceinline__ v8f wmma_f16(v16h a, v16h b, v8f c) {
    return __builtin_amdgcn_wmma_f32_16x16x32_f16(false, a, false, b,
                                                  (short)0, c, false, false);
}

// ---------------------------------------------------------------------------
// TDM: issue a 2D tensor tile load Global -> LDS (f16 elements).
// Builds D# group0/group1 per CDNA5 ISA 8.3/8.4 and issues the 2-group form
// (VADDR2/3 = NULL => tensors up to 2D). Tracked by TENSORcnt.
//   group0: [1:0]=count=1, [63:32]=lds byte addr, [120:64]=global addr,
//           [127:126]=type=2
//   group1: [17:16]=data_size=1 (2B), tensor_dim0/1, tile_dim0/1,
//           tensor_dim0_stride (elements)
// ---------------------------------------------------------------------------
__device__ __forceinline__ void tdm_load_2d(unsigned lds_addr,
                                            const _Float16* gsrc,
                                            unsigned tensor_d0, unsigned tensor_d1,
                                            unsigned tile_d0, unsigned tile_d1,
                                            unsigned stride0_elems) {
    unsigned long long ga = (unsigned long long)(uintptr_t)gsrc;
    u32x4 g0;
    g0[0] = 1u;                                        // count=1, user load
    g0[1] = lds_addr;                                  // LDS byte address
    g0[2] = (unsigned)(ga & 0xffffffffu);              // global_addr[31:0]
    g0[3] = (unsigned)((ga >> 32) & 0x1ffffffu)        // global_addr[56:32]
          | (2u << 30);                                // type = 2 ("image")
    u32x8 g1;
    g1[0] = (1u << 16);                                // data_size=1 -> 2 bytes
    g1[1] = (tensor_d0 & 0xffffu) << 16;               // tensor_dim0[15:0]
    g1[2] = ((tensor_d0 >> 16) & 0xffffu)              // tensor_dim0[31:16]
          | ((tensor_d1 & 0xffffu) << 16);             // tensor_dim1[15:0]
    g1[3] = ((tensor_d1 >> 16) & 0xffffu)              // tensor_dim1[31:16]
          | ((tile_d0 & 0xffffu) << 16);               // tile_dim0
    g1[4] = (tile_d1 & 0xffffu);                       // tile_dim1 (tile_dim2=0)
    g1[5] = stride0_elems;                             // tensor_dim0_stride[31:0]
    g1[6] = 0;                                         // stride0 hi / dim1_stride lo
    g1[7] = 0;
    asm volatile("tensor_load_to_lds %0, %1" :: "s"(g0), "s"(g1) : "memory");
}

__device__ __forceinline__ unsigned lds_off(const void* p) {
    return (unsigned)(unsigned long long)(uintptr_t)p;  // low 32 bits = LDS offset
}

// ---------------------------------------------------------------------------
// 1) LayerNorm: x[ROWS][DIM] f32 -> xn f16. One 256-thread block per row.
// ---------------------------------------------------------------------------
__global__ void ln_kernel(const float* __restrict__ x,
                          const float* __restrict__ gamma,
                          const float* __restrict__ beta,
                          _Float16* __restrict__ xn) {
    __shared__ float red[256];
    int row = blockIdx.x;
    int t = threadIdx.x;
    const float* xr = x + (size_t)row * DIM;
    float v0 = xr[t], v1 = xr[t + 256];

    red[t] = v0 + v1;
    __syncthreads();
    for (int off = 128; off > 0; off >>= 1) {
        if (t < off) red[t] += red[t + off];
        __syncthreads();
    }
    float mu = red[0] * (1.0f / DIM);
    __syncthreads();

    float d0 = v0 - mu, d1 = v1 - mu;
    red[t] = d0 * d0 + d1 * d1;
    __syncthreads();
    for (int off = 128; off > 0; off >>= 1) {
        if (t < off) red[t] += red[t + off];
        __syncthreads();
    }
    float inv = rsqrtf(red[0] * (1.0f / DIM) + 1e-5f);

    _Float16* xo = xn + (size_t)row * DIM;
    xo[t]       = (_Float16)(d0 * inv * gamma[t]       + beta[t]);
    xo[t + 256] = (_Float16)(d1 * inv * gamma[t + 256] + beta[t + 256]);
}

// ---------------------------------------------------------------------------
// 2) Transpose + f32->f16: src[R][C] -> dst[C][R] (B-frag contiguous along K)
// ---------------------------------------------------------------------------
__global__ void transpose_f16(const float* __restrict__ src,
                              _Float16* __restrict__ dst, int R, int C) {
    int idx = blockIdx.x * blockDim.x + threadIdx.x;
    if (idx >= R * C) return;
    int r = idx / C, c = idx % C;
    dst[(size_t)c * R + r] = (_Float16)src[idx];
}

// ---------------------------------------------------------------------------
// 3) QKV GEMM: one wave per 32x64 tile (2 A-frags, 8 accum, 8 WMMA/k-step).
//    Scatters into q (pre-scaled), k (row-major [b][h][n][d]) and
//    v_t (transposed [b][h][d][n]).
// ---------------------------------------------------------------------------
__global__ void qkv_gemm(const _Float16* __restrict__ xn,
                         const _Float16* __restrict__ wt,   // [QKVN][DIM]
                         _Float16* __restrict__ q,
                         _Float16* __restrict__ kmat,
                         _Float16* __restrict__ v_t) {
    int lane = threadIdx.x;
    int m0 = blockIdx.x * 32;
    int n0 = blockIdx.y * 64;
    int nl = lane & 15;

    v8f acc[2][4] = {};
    const _Float16* arow0 = xn + (size_t)(m0 + nl) * DIM;
    const _Float16* arow1 = arow0 + (size_t)16 * DIM;

    for (int k0 = 0; k0 < DIM; k0 += 32) {
        v16h a0 = load_frag(arow0 + k0);
        v16h a1 = load_frag(arow1 + k0);
#pragma unroll
        for (int t = 0; t < 4; ++t) {
            v16h bf = load_frag(wt + (size_t)(n0 + t * 16 + nl) * DIM + k0);
            acc[0][t] = wmma_f16(a0, bf, acc[0][t]);
            acc[1][t] = wmma_f16(a1, bf, acc[1][t]);
        }
    }

    int hi = (lane & 16) ? 8 : 0;
#pragma unroll
    for (int half = 0; half < 2; ++half) {
#pragma unroll
        for (int t = 0; t < 4; ++t) {
            int col = n0 + t * 16 + nl;
            int which = col / DIM;           // 0=q, 1=k, 2=v (uniform per tile)
            int c = col % DIM;
            int h = c / DHEAD, d = c % DHEAD;
#pragma unroll
            for (int r = 0; r < 8; ++r) {
                int grow = m0 + half * 16 + r + hi;
                int b = grow >> 12;
                int n = grow & (NTOK - 1);
                size_t bh = (size_t)b * HEADS + h;
                float val = acc[half][t][r];
                if (which == 0)
                    q[(bh * NTOK + n) * DHEAD + d] = (_Float16)(val * SCALE);
                else if (which == 1)
                    kmat[(bh * NTOK + n) * DHEAD + d] = (_Float16)val;
                else
                    v_t[(bh * DHEAD + d) * NTOK + n] = (_Float16)val;
            }
        }
    }
}

// ---------------------------------------------------------------------------
// 4) Flash attention, TDM-fed: 4 waves/block, each owning a 16-row Q tile
//    (64 rows/block). Wave 0 drives the Tensor Data Mover to double-buffer
//    32-key K tiles (32x64) and V^T tiles (64x32) into LDS; all waves
//    consume them through ds_load fragments. Softmax is online, with
//    half-wave shfl_xor row reductions; P bounces through per-wave LDS
//    (same-wave DS ops are in-order, no barrier needed).
// ---------------------------------------------------------------------------
__global__ void flash_attn(const _Float16* __restrict__ q,
                           const _Float16* __restrict__ kmat,
                           const _Float16* __restrict__ v_t,
                           _Float16* __restrict__ aout) {
    __shared__ __align__(16) _Float16 ldsK[2][KSTEP * DHEAD];   // 2 x 4 KB
    __shared__ __align__(16) _Float16 ldsV[2][DHEAD * KSTEP];   // 2 x 4 KB
    __shared__ __align__(16) _Float16 ldsP[4][16 * 32];         // 4 x 1 KB

    int tid  = threadIdx.x;
    int lane = tid & 31;
    int wave = tid >> 5;
    int h = blockIdx.y, b = blockIdx.z;
    size_t bh = (size_t)b * HEADS + h;
    int qrow0 = blockIdx.x * 64 + wave * 16;
    int hi = (lane >> 4) & 1;
    int nl = lane & 15;

    const _Float16* kbase = kmat + bh * NTOK * DHEAD;   // [4096][64]
    const _Float16* vbase = v_t + bh * DHEAD * NTOK;    // [64][4096]

    const _Float16* qbase = q + (bh * NTOK + qrow0 + nl) * DHEAD;
    v16h qa0 = load_frag(qbase);        // d = 0..31
    v16h qa1 = load_frag(qbase + 32);   // d = 32..63

    float m_i[8], l_i[8];
    v8f acc[4] = {};
#pragma unroll
    for (int r = 0; r < 8; ++r) { m_i[r] = -__builtin_inff(); l_i[r] = 0.f; }

    constexpr int NST = NTOK / KSTEP;   // 128 stages

    if (wave == 0) {
        // prologue: stage 0 into buffer 0
        tdm_load_2d(lds_off(&ldsK[0][0]), kbase,
                    DHEAD, NTOK, DHEAD, KSTEP, DHEAD);
        tdm_load_2d(lds_off(&ldsV[0][0]), vbase,
                    NTOK, DHEAD, KSTEP, DHEAD, NTOK);
    }

    for (int st = 0; st < NST; ++st) {
        int buf = st & 1;
        if (wave == 0) {
            if (st + 1 < NST) {
                const _Float16* kn = kbase + (size_t)(st + 1) * KSTEP * DHEAD;
                const _Float16* vn = vbase + (size_t)(st + 1) * KSTEP;
                unsigned remain = NTOK - (st + 1) * KSTEP;
                tdm_load_2d(lds_off(&ldsK[buf ^ 1][0]), kn,
                            DHEAD, remain, DHEAD, KSTEP, DHEAD);
                tdm_load_2d(lds_off(&ldsV[buf ^ 1][0]), vn,
                            remain, DHEAD, KSTEP, DHEAD, NTOK);
                __builtin_amdgcn_s_wait_tensorcnt(2);   // stage st complete
            } else {
                __builtin_amdgcn_s_wait_tensorcnt(0);
            }
        }
        __syncthreads();    // K/V tiles for stage st visible to all waves

        // --- scores: S[16 x 32] via chained WMMAs from LDS K tile ---
        const _Float16* kt = &ldsK[buf][0];
        v8f s0 = {}, s1 = {};
        s0 = wmma_f16(qa0, load_frag(kt + (size_t)nl * DHEAD), s0);
        s0 = wmma_f16(qa1, load_frag(kt + (size_t)nl * DHEAD + 32), s0);
        s1 = wmma_f16(qa0, load_frag(kt + (size_t)(16 + nl) * DHEAD), s1);
        s1 = wmma_f16(qa1, load_frag(kt + (size_t)(16 + nl) * DHEAD + 32), s1);

        // --- online softmax (row m = r + hi*8 lives in one 16-lane half) ---
        _Float16* pw = &ldsP[wave][0];
#pragma unroll
        for (int r = 0; r < 8; ++r) {
            float mx = fmaxf(s0[r], s1[r]);
#pragma unroll
            for (int off = 1; off < 16; off <<= 1)
                mx = fmaxf(mx, __shfl_xor(mx, off, 32));
            float mnew = fmaxf(m_i[r], mx);
            float p0 = __expf(s0[r] - mnew);
            float p1 = __expf(s1[r] - mnew);
            float rs = p0 + p1;
#pragma unroll
            for (int off = 1; off < 16; off <<= 1)
                rs += __shfl_xor(rs, off, 32);
            float sc = __expf(m_i[r] - mnew);
            l_i[r] = l_i[r] * sc + rs;
            m_i[r] = mnew;
#pragma unroll
            for (int t = 0; t < 4; ++t) acc[t][r] *= sc;
            int m = r + hi * 8;
            pw[m * 32 + nl]      = (_Float16)p0;
            pw[m * 32 + 16 + nl] = (_Float16)p1;
        }

        // --- O += P @ V_block (P re-read in A-layout; same-wave DS in-order)
        v16h pa = load_frag(pw + nl * 32);
#pragma unroll
        for (int t = 0; t < 4; ++t) {
            const _Float16* vb = &ldsV[buf][(t * 16 + nl) * KSTEP];
            acc[t] = wmma_f16(pa, load_frag(vb), acc[t]);
        }
        __syncthreads();    // all waves done with buf before TDM refills it
    }

#pragma unroll
    for (int r = 0; r < 8; ++r) {
        float inv = 1.0f / l_i[r];
        size_t grow = (size_t)b * NTOK + qrow0 + r + hi * 8;
#pragma unroll
        for (int t = 0; t < 4; ++t) {
            int col = h * DHEAD + t * 16 + nl;
            aout[grow * DIM + col] = (_Float16)(acc[t][r] * inv);
        }
    }
}

// ---------------------------------------------------------------------------
// 5) Output projection: aout[ROWS][DIM] @ w_out + b_out -> out f32.
//    One wave per 32x64 tile.
// ---------------------------------------------------------------------------
__global__ void out_gemm(const _Float16* __restrict__ aout,
                         const _Float16* __restrict__ wt,   // [DIM out][DIM in]
                         const float* __restrict__ bias,
                         float* __restrict__ out) {
    int lane = threadIdx.x;
    int m0 = blockIdx.x * 32;
    int n0 = blockIdx.y * 64;
    int nl = lane & 15;

    v8f acc[2][4] = {};
    const _Float16* arow0 = aout + (size_t)(m0 + nl) * DIM;
    const _Float16* arow1 = arow0 + (size_t)16 * DIM;
    for (int k0 = 0; k0 < DIM; k0 += 32) {
        v16h a0 = load_frag(arow0 + k0);
        v16h a1 = load_frag(arow1 + k0);
#pragma unroll
        for (int t = 0; t < 4; ++t) {
            v16h bf = load_frag(wt + (size_t)(n0 + t * 16 + nl) * DIM + k0);
            acc[0][t] = wmma_f16(a0, bf, acc[0][t]);
            acc[1][t] = wmma_f16(a1, bf, acc[1][t]);
        }
    }

    int hi = (lane & 16) ? 8 : 0;
#pragma unroll
    for (int half = 0; half < 2; ++half) {
#pragma unroll
        for (int t = 0; t < 4; ++t) {
            int col = n0 + t * 16 + nl;
            float bv = bias[col];
#pragma unroll
            for (int r = 0; r < 8; ++r)
                out[(size_t)(m0 + half * 16 + r + hi) * DIM + col] =
                    acc[half][t][r] + bv;
        }
    }
}

// ---------------------------------------------------------------------------
extern "C" void kernel_launch(void* const* d_in, const int* in_sizes, int n_in,
                              void* d_out, int out_size, void* d_ws, size_t ws_size,
                              hipStream_t stream) {
    const float* x     = (const float*)d_in[0];
    const float* gamma = (const float*)d_in[1];
    const float* beta  = (const float*)d_in[2];
    const float* w_qkv = (const float*)d_in[3];
    const float* w_out = (const float*)d_in[4];
    const float* b_out = (const float*)d_in[5];
    float* out = (float*)d_out;

    char* ws = (char*)d_ws;
    constexpr size_t XN_B    = (size_t)ROWS * DIM * 2;                   // 8 MB
    constexpr size_t WQKVT_B = (size_t)QKVN * DIM * 2;                   // 1.5 MB
    constexpr size_t WOUTT_B = (size_t)DIM * DIM * 2;                    // 0.5 MB
    constexpr size_t QKV_B   = (size_t)BATCH * HEADS * NTOK * DHEAD * 2; // 8 MB

    _Float16* xn    = (_Float16*)(ws);
    _Float16* wqkvt = (_Float16*)(ws + XN_B);
    _Float16* woutt = (_Float16*)(ws + XN_B + WQKVT_B);
    _Float16* qbuf  = (_Float16*)(ws + XN_B + WQKVT_B + WOUTT_B);
    _Float16* kbuf  = (_Float16*)(ws + XN_B + WQKVT_B + WOUTT_B + QKV_B);
    _Float16* vtbuf = (_Float16*)(ws + XN_B + WQKVT_B + WOUTT_B + 2 * QKV_B);
    _Float16* abuf  = (_Float16*)(ws + XN_B + WQKVT_B + WOUTT_B + 3 * QKV_B);

    ln_kernel<<<ROWS, 256, 0, stream>>>(x, gamma, beta, xn);

    transpose_f16<<<(DIM * QKVN + 255) / 256, 256, 0, stream>>>(w_qkv, wqkvt, DIM, QKVN);
    transpose_f16<<<(DIM * DIM + 255) / 256, 256, 0, stream>>>(w_out, woutt, DIM, DIM);

    qkv_gemm<<<dim3(ROWS / 32, QKVN / 64), 32, 0, stream>>>(xn, wqkvt, qbuf, kbuf, vtbuf);

    flash_attn<<<dim3(NTOK / 64, HEADS, BATCH), 128, 0, stream>>>(qbuf, kbuf, vtbuf, abuf);

    out_gemm<<<dim3(ROWS / 32, DIM / 64), 32, 0, stream>>>(abuf, woutt, b_out, out);
}
